// AttnDecoderRNN_10788957848009
// MI455X (gfx1250) — matrix operand order repaired and checked by reading, not compiled, not run on hardware
//
#include <hip/hip_runtime.h>
#include <hip/hip_bf16.h>

// Problem constants (AttnDecoderRNN step)
constexpr int B_ = 256;
constexpr int S_ = 15;
constexpr int H_ = 512;
constexpr int V_ = 50257;

typedef __attribute__((ext_vector_type(16))) __bf16 v16bf;
typedef __attribute__((ext_vector_type(8)))  float  v8f;

// ---------- native bf16 conversion (hardware cvt, RNE) ----------
__device__ __forceinline__ void cvt_hilo(float f, unsigned short& hi, unsigned short& lo) {
    __bf16 h = (__bf16)f;
    hi = __builtin_bit_cast(unsigned short, h);
    __bf16 l = (__bf16)(f - (float)h);
    lo = __builtin_bit_cast(unsigned short, l);
}

union AU {
    uint4          u[2];
    unsigned short s[16];
    v16bf          v;
};

// =====================================================================
// Kernel 1: u[b,:] = h[b,:] @ attn_W   ([B,H] @ [H,H], W row-major [o,h'])
// grid (H/128, B/16), 256 threads. Coalesced W reads (thread -> column j).
// =====================================================================
__global__ __launch_bounds__(256) void attn_u_kernel(
    const float* __restrict__ hidden, const float* __restrict__ attnW,
    float* __restrict__ u) {
    __shared__ float hs[16 * H_];
    const int t = threadIdx.x;
    const int rowBase = blockIdx.y * 16;
    const int j = blockIdx.x * 128 + (t & 127);
    const int rhalf = t >> 7;
    for (int i = t; i < 16 * H_; i += 256)
        hs[i] = hidden[rowBase * H_ + i];
    __syncthreads();
    float acc[8] = {};
    for (int o = 0; o < H_; ++o) {
        const float w = attnW[o * H_ + j];
#pragma unroll
        for (int r = 0; r < 8; ++r)
            acc[r] += hs[(rhalf * 8 + r) * H_ + o] * w;
    }
#pragma unroll
    for (int r = 0; r < 8; ++r)
        u[(rowBase + rhalf * 8 + r) * H_ + j] = acc[r];
}

// =====================================================================
// Kernel 2: scores[b,s] = enc[b,s]·u[b] + h[b]·attn_b ; softmax over S;
//           attn_weights out ; context[b,:] = sum_s w_s * enc[b,s,:]
// grid (B), 128 threads.
// =====================================================================
__global__ __launch_bounds__(128) void attn_ctx_kernel(
    const float* __restrict__ hidden, const float* __restrict__ enc,
    const float* __restrict__ attnb, const float* __restrict__ u,
    float* __restrict__ attw, float* __restrict__ ctx) {
    __shared__ float es[S_ * H_];
    __shared__ float red[128];
    __shared__ float part[16 * 8];
    __shared__ float sc[16];
    __shared__ float wts[16];
    __shared__ float mxsm[2];
    const int b = blockIdx.x, t = threadIdx.x;
    for (int i = t; i < S_ * H_; i += 128) es[i] = enc[b * S_ * H_ + i];
    float p = 0.f;
    for (int k = t; k < H_; k += 128) p += hidden[b * H_ + k] * attnb[k];
    red[t] = p;
    __syncthreads();
    for (int off = 64; off > 0; off >>= 1) {
        if (t < off) red[t] += red[t + off];
        __syncthreads();
    }
    const float hb = red[0];
    const int s = t & 15, chunk = t >> 4;
    float ps = 0.f;
    if (s < S_) {
        const float* er = es + s * H_ + chunk * 64;
        const float* ur = u + b * H_ + chunk * 64;
        for (int k = 0; k < 64; ++k) ps += er[k] * ur[k];
    }
    part[s * 8 + chunk] = ps;
    __syncthreads();
    if (t < S_) {
        float v = hb;
#pragma unroll
        for (int c2 = 0; c2 < 8; ++c2) v += part[t * 8 + c2];
        sc[t] = v;
    }
    __syncthreads();
    if (t == 0) {
        float m = sc[0];
        for (int i2 = 1; i2 < S_; ++i2) m = fmaxf(m, sc[i2]);
        float sm = 0.f;
        for (int i2 = 0; i2 < S_; ++i2) sm += expf(sc[i2] - m);
        mxsm[0] = m; mxsm[1] = sm;
    }
    __syncthreads();
    if (t < S_) {
        float w = expf(sc[t] - mxsm[0]) / mxsm[1];
        wts[t] = w;
        attw[b * S_ + t] = w;
    }
    __syncthreads();
    for (int k = t; k < H_; k += 128) {
        float acc = 0.f;
#pragma unroll
        for (int s2 = 0; s2 < S_; ++s2) acc += wts[s2] * es[s2 * H_ + k];
        ctx[b * H_ + k] = acc;
    }
}

// =====================================================================
// Kernel 3: fused GRU step. Each block: 16 batch rows x 64 hidden cols.
// grid (H/64, B/16), 256 threads. Single 32KB LDS buffer reused 3x.
// =====================================================================
__global__ __launch_bounds__(256) void gru_kernel(
    const int* __restrict__ tok, const float* __restrict__ hidden,
    const float* __restrict__ emb, const float* __restrict__ Wih,
    const float* __restrict__ Whh, const float* __restrict__ bih,
    const float* __restrict__ bhh, const float* __restrict__ ctx,
    float* __restrict__ hnew) {
    __shared__ float S1[16 * H_];
    __shared__ int tk[16];
    const int t = threadIdx.x;
    const int c = t & 63, rq = t >> 6;
    const int rowBase = blockIdx.y * 16;
    const int cb = blockIdx.x * 64;
    const int jr = cb + c, jz = jr + H_, jn = jr + 2 * H_;
    if (t < 16) tk[t] = tok[rowBase + t];
    __syncthreads();
    for (int i = t; i < 16 * H_; i += 256) {
        int m = i >> 9, k = i & (H_ - 1);
        S1[i] = emb[(size_t)tk[m] * H_ + k];
    }
    __syncthreads();
    float air[4] = {}, aiz[4] = {}, ain[4] = {};
    float ahr[4] = {}, ahz[4] = {}, ahn[4] = {};
    const float* wr0 = Wih + (size_t)jr * (2 * H_);
    const float* wz0 = Wih + (size_t)jz * (2 * H_);
    const float* wn0 = Wih + (size_t)jn * (2 * H_);
    for (int k = 0; k < H_; ++k) {
        const float wr = wr0[k], wz = wz0[k], wn = wn0[k];
#pragma unroll
        for (int q = 0; q < 4; ++q) {
            const float x = S1[(rq * 4 + q) * H_ + k];
            air[q] += x * wr; aiz[q] += x * wz; ain[q] += x * wn;
        }
    }
    __syncthreads();
    for (int i = t; i < 16 * H_; i += 256) {
        int m = i >> 9, k = i & (H_ - 1);
        S1[i] = ctx[(rowBase + m) * H_ + k];
    }
    __syncthreads();
    for (int k = 0; k < H_; ++k) {
        const float wr = wr0[H_ + k], wz = wz0[H_ + k], wn = wn0[H_ + k];
#pragma unroll
        for (int q = 0; q < 4; ++q) {
            const float x = S1[(rq * 4 + q) * H_ + k];
            air[q] += x * wr; aiz[q] += x * wz; ain[q] += x * wn;
        }
    }
    __syncthreads();
    for (int i = t; i < 16 * H_; i += 256) {
        int m = i >> 9, k = i & (H_ - 1);
        S1[i] = hidden[(rowBase + m) * H_ + k];
    }
    __syncthreads();
    const float* vr0 = Whh + (size_t)jr * H_;
    const float* vz0 = Whh + (size_t)jz * H_;
    const float* vn0 = Whh + (size_t)jn * H_;
    for (int k = 0; k < H_; ++k) {
        const float wr = vr0[k], wz = vz0[k], wn = vn0[k];
#pragma unroll
        for (int q = 0; q < 4; ++q) {
            const float x = S1[(rq * 4 + q) * H_ + k];
            ahr[q] += x * wr; ahz[q] += x * wz; ahn[q] += x * wn;
        }
    }
    const float bir = bih[jr], biz = bih[jz], bin = bih[jn];
    const float bhr = bhh[jr], bhz = bhh[jz], bhn = bhh[jn];
#pragma unroll
    for (int q = 0; q < 4; ++q) {
        const int m = rq * 4 + q;
        const float r = 1.f / (1.f + expf(-(air[q] + bir + ahr[q] + bhr)));
        const float z = 1.f / (1.f + expf(-(aiz[q] + biz + ahz[q] + bhz)));
        const float n = tanhf(ain[q] + bin + r * (ahn[q] + bhn));
        const float ho = S1[m * H_ + jr];
        hnew[(rowBase + m) * H_ + jr] = (1.f - z) * n + z * ho;
    }
}

// =====================================================================
// Kernel 3b: pre-convert h_new -> bf16 hi/lo planes (done once; tiny)
// =====================================================================
__global__ __launch_bounds__(256) void cvt_hnew_kernel(
    const float* __restrict__ hnew, unsigned short* __restrict__ Ahig,
    unsigned short* __restrict__ Alog) {
    const int i = blockIdx.x * 256 + threadIdx.x;
    if (i < B_ * H_) {
        unsigned short hi, lo;
        cvt_hilo(hnew[i], hi, lo);
        Ahig[i] = hi;
        Alog[i] = lo;
    }
}

// =====================================================================
// Kernel 4: logits = h_new @ out_W^T + out_b, WMMA bf16 with hi/lo split.
// Block tile: 256(M) x 64(N); 8 waves, each wave = 2 M-tiles x 4 N-tiles.
// out_W read from HBM exactly once. K chunks of 32:
//   - A chunk staged via global_load_async_to_lds_b128 (ASYNCcnt DMA)
//   - B chunk: software-pipelined float4 loads (preloaded across WMMAs),
//     converted fp32->bf16 hi/lo with hardware cvt, stored to LDS.
// grid (ceil(V/64)), 256 threads.
// =====================================================================
constexpr int NB_  = 64;   // cols per block
constexpr int KC_  = 32;   // k chunk
constexpr int LDH_ = 40;   // padded LDS row stride (halfs): conflict-free b128

__global__ __launch_bounds__(256) void logits_kernel(
    const unsigned short* __restrict__ Ahig, const unsigned short* __restrict__ Alog,
    const float* __restrict__ outW, const float* __restrict__ outb,
    float* __restrict__ logits) {
    __shared__ unsigned short sAhi[256 * LDH_];  // 20 KB
    __shared__ unsigned short sAlo[256 * LDH_];  // 20 KB
    __shared__ unsigned short sBhi[NB_ * LDH_];  // 5 KB
    __shared__ unsigned short sBlo[NB_ * LDH_];  // 5 KB
    const int t = threadIdx.x;
    const int cb = blockIdx.x * NB_;
    const int wave = t >> 5, lane = t & 31;
    const int mlane = lane & 15;
    const int hiHalf = lane >> 4;  // 0: lanes 0-15, 1: lanes 16-31

    // ---- B staging mapping: this thread owns (colr0,seg0) and (colr1,seg0) ----
    const int seg0 = t & 7;          // float4 segment within 32-k chunk
    const int colr0 = t >> 3;        // 0..31
    const int colr1 = colr0 + 32;    // 32..63
    const int col0 = cb + colr0, col1 = cb + colr1;
    const size_t boff0 = (size_t)(col0 < V_ ? col0 : V_ - 1) * H_ + seg0 * 4;
    const size_t boff1 = (size_t)(col1 < V_ ? col1 : V_ - 1) * H_ + seg0 * 4;
    // prologue: preload B chunk for kb = 0
    float4 nw0 = *(const float4*)(outW + boff0);
    float4 nw1 = *(const float4*)(outW + boff1);

    // ---- A async staging mapping ----
    const unsigned ldsAhi = (unsigned)(size_t)&sAhi[0];
    const unsigned ldsAlo = (unsigned)(size_t)&sAlo[0];
    const int arow = t >> 2;   // 0..63 (row within 64-row group)
    const int aseg = t & 3;    // 8-half segment

    v8f acc[2][4] = {};

    for (int kb = 0; kb < H_; kb += KC_) {
        // ---- issue async DMA: A chunk (256 rows x 32 halfs, both planes) ----
#pragma unroll
        for (int p = 0; p < 4; ++p) {
            const int row = p * 64 + arow;
            const unsigned go = (unsigned)((row * H_ + kb + aseg * 8) * 2);
            const unsigned lo = (unsigned)((row * LDH_ + aseg * 8) * 2);
            asm volatile("global_load_async_to_lds_b128 %0, %1, %2"
                         :: "v"(ldsAhi + lo), "v"(go), "s"(Ahig) : "memory");
            asm volatile("global_load_async_to_lds_b128 %0, %1, %2"
                         :: "v"(ldsAlo + lo), "v"(go), "s"(Alog) : "memory");
        }
        // ---- convert preloaded B chunk -> LDS (overlaps with async DMA) ----
        {
            unsigned short h0, h1, h2, h3, l0, l1, l2, l3;
            cvt_hilo(nw0.x, h0, l0); cvt_hilo(nw0.y, h1, l1);
            cvt_hilo(nw0.z, h2, l2); cvt_hilo(nw0.w, h3, l3);
            unsigned int* dh = (unsigned int*)(sBhi + colr0 * LDH_ + seg0 * 4);
            dh[0] = (unsigned)h0 | ((unsigned)h1 << 16);
            dh[1] = (unsigned)h2 | ((unsigned)h3 << 16);
            unsigned int* dl = (unsigned int*)(sBlo + colr0 * LDH_ + seg0 * 4);
            dl[0] = (unsigned)l0 | ((unsigned)l1 << 16);
            dl[1] = (unsigned)l2 | ((unsigned)l3 << 16);
            cvt_hilo(nw1.x, h0, l0); cvt_hilo(nw1.y, h1, l1);
            cvt_hilo(nw1.z, h2, l2); cvt_hilo(nw1.w, h3, l3);
            dh = (unsigned int*)(sBhi + colr1 * LDH_ + seg0 * 4);
            dh[0] = (unsigned)h0 | ((unsigned)h1 << 16);
            dh[1] = (unsigned)h2 | ((unsigned)h3 << 16);
            dl = (unsigned int*)(sBlo + colr1 * LDH_ + seg0 * 4);
            dl[0] = (unsigned)l0 | ((unsigned)l1 << 16);
            dl[1] = (unsigned)l2 | ((unsigned)l3 << 16);
        }
        asm volatile("s_wait_asynccnt 0x0" ::: "memory");
        __syncthreads();
        // ---- B fragments (shared across both M-tiles) ----
        AU bh[4], bl[4];
#pragma unroll
        for (int n = 0; n < 4; ++n) {
            const unsigned short* bp = sBhi + (n * 16 + mlane) * LDH_ + hiHalf * 16;
            bh[n].u[0] = ((const uint4*)bp)[0];
            bh[n].u[1] = ((const uint4*)bp)[1];
            const unsigned short* bq = sBlo + (n * 16 + mlane) * LDH_ + hiHalf * 16;
            bl[n].u[0] = ((const uint4*)bq)[0];
            bl[n].u[1] = ((const uint4*)bq)[1];
        }
        // ---- A fragments ----
        AU ah[2], al[2];
#pragma unroll
        for (int mt = 0; mt < 2; ++mt) {
            const int row = wave * 32 + mt * 16 + mlane;
            const unsigned short* ap = sAhi + row * LDH_ + hiHalf * 16;
            const unsigned short* aq = sAlo + row * LDH_ + hiHalf * 16;
            ah[mt].u[0] = ((const uint4*)ap)[0];
            ah[mt].u[1] = ((const uint4*)ap)[1];
            al[mt].u[0] = ((const uint4*)aq)[0];
            al[mt].u[1] = ((const uint4*)aq)[1];
        }
        // ---- software pipeline: preload next B chunk before the WMMA block ----
        if (kb + KC_ < H_) {
            nw0 = *(const float4*)(outW + boff0 + kb + KC_);
            nw1 = *(const float4*)(outW + boff1 + kb + KC_);
        }
        // ---- 24 WMMAs ----
#pragma unroll
        for (int mt = 0; mt < 2; ++mt) {
#pragma unroll
            for (int n = 0; n < 4; ++n) {
                acc[mt][n] = __builtin_amdgcn_wmma_f32_16x16x32_bf16(
                    false, ah[mt].v, false, bh[n].v, (short)0, acc[mt][n], false, false);
                acc[mt][n] = __builtin_amdgcn_wmma_f32_16x16x32_bf16(
                    false, ah[mt].v, false, bl[n].v, (short)0, acc[mt][n], false, false);
                acc[mt][n] = __builtin_amdgcn_wmma_f32_16x16x32_bf16(
                    false, al[mt].v, false, bh[n].v, (short)0, acc[mt][n], false, false);
            }
        }
        __syncthreads();
    }
    // ---- epilogue: C layout vgpr r -> M=r (+8 for lanes 16-31) ----
#pragma unroll
    for (int mt = 0; mt < 2; ++mt) {
        const int row0 = wave * 32 + mt * 16 + hiHalf * 8;
#pragma unroll
        for (int n = 0; n < 4; ++n) {
            const int col = cb + n * 16 + mlane;
            if (col < V_) {
                const float bias = outb[col];
#pragma unroll
                for (int r = 0; r < 8; ++r)
                    logits[(size_t)(row0 + r) * V_ + col] = acc[mt][n][r] + bias;
            }
        }
    }
}

// =====================================================================
// Kernel 5: per-row online logsumexp over V. grid (B), 256 threads.
// =====================================================================
__global__ __launch_bounds__(256) void lse_kernel(
    const float* __restrict__ lg, float* __restrict__ lse) {
    __shared__ float ms[256], ss[256];
    const int b = blockIdx.x, t = threadIdx.x;
    float m = -3.4e38f, s = 0.f;
    for (int i = t; i < V_; i += 256) {
        const float x = lg[(size_t)b * V_ + i];
        const float nm = fmaxf(m, x);
        s = s * expf(m - nm) + expf(x - nm);
        m = nm;
    }
    ms[t] = m; ss[t] = s;
    __syncthreads();
    for (int off = 128; off > 0; off >>= 1) {
        if (t < off) {
            const float m2 = ms[t + off], s2 = ss[t + off];
            const float nm = fmaxf(ms[t], m2);
            ss[t] = ss[t] * expf(ms[t] - nm) + s2 * expf(m2 - nm);
            ms[t] = nm;
        }
        __syncthreads();
    }
    if (t == 0) lse[b] = ms[0] + logf(ss[0]);
}

// =====================================================================
// Kernel 6: log_probs = logits - lse[row], in place.
// =====================================================================
__global__ __launch_bounds__(256) void logp_kernel(
    float* __restrict__ lg, const float* __restrict__ lse) {
    const int col = blockIdx.x * 256 + threadIdx.x;
    const int row = blockIdx.y;
    if (col < V_) lg[(size_t)row * V_ + col] -= lse[row];
}

extern "C" void kernel_launch(void* const* d_in, const int* in_sizes, int n_in,
                              void* d_out, int out_size, void* d_ws, size_t ws_size,
                              hipStream_t stream) {
    const int*   tok    = (const int*)  d_in[0];
    const float* hidden = (const float*)d_in[1];   // [1,B,H]
    const float* enc    = (const float*)d_in[2];   // [B,S,H]
    const float* emb    = (const float*)d_in[3];   // [V,H]
    const float* attnW  = (const float*)d_in[4];   // [H,H]
    const float* attnb  = (const float*)d_in[5];   // [H]
    const float* Wih    = (const float*)d_in[6];   // [3H,2H]
    const float* Whh    = (const float*)d_in[7];   // [3H,H]
    const float* bih    = (const float*)d_in[8];   // [3H]
    const float* bhh    = (const float*)d_in[9];   // [3H]
    const float* outW   = (const float*)d_in[10];  // [V,H]
    const float* outb   = (const float*)d_in[11];  // [V]

    float* out_logp = (float*)d_out;                       // [B,1,V]
    float* out_hnew = out_logp + (size_t)B_ * V_;          // [1,B,H]
    float* out_attw = out_hnew + (size_t)B_ * H_;          // [B,S]

    float* ws_u   = (float*)d_ws;                          // [B,H] f32
    float* ws_ctx = ws_u + (size_t)B_ * H_;                // [B,H] f32
    float* ws_lse = ws_ctx + (size_t)B_ * H_;              // [B]   f32
    unsigned short* ws_ahi = (unsigned short*)(ws_lse + B_);          // [B,H] bf16
    unsigned short* ws_alo = ws_ahi + (size_t)B_ * H_;                // [B,H] bf16

    attn_u_kernel<<<dim3(H_ / 128, B_ / 16), 256, 0, stream>>>(hidden, attnW, ws_u);
    attn_ctx_kernel<<<B_, 128, 0, stream>>>(hidden, enc, attnb, ws_u, out_attw, ws_ctx);
    gru_kernel<<<dim3(H_ / 64, B_ / 16), 256, 0, stream>>>(tok, hidden, emb, Wih, Whh,
                                                           bih, bhh, ws_ctx, out_hnew);
    cvt_hnew_kernel<<<(B_ * H_ + 255) / 256, 256, 0, stream>>>(out_hnew, ws_ahi, ws_alo);
    logits_kernel<<<(V_ + NB_ - 1) / NB_, 256, 0, stream>>>(ws_ahi, ws_alo, outW, outb,
                                                            out_logp);
    lse_kernel<<<B_, 256, 0, stream>>>(out_logp, ws_lse);
    logp_kernel<<<dim3((V_ + 255) / 256, B_), 256, 0, stream>>>(out_logp, ws_lse);
}
